// InternA1_31533649887536
// MI455X (gfx1250) — matrix operand order repaired and checked by reading, not compiled, not run on hardware
//
#include <hip/hip_runtime.h>
#include <hip/hip_bf16.h>
#include <math.h>

typedef __bf16 bf16_t;
typedef __attribute__((ext_vector_type(8)))  bf16_t v8bf;
typedef __attribute__((ext_vector_type(16))) bf16_t v16bf;
typedef __attribute__((ext_vector_type(8)))  float  v8f;

#define HID   896
#define NH    14
#define NKV   2
#define HD    64
#define INTER 4864
#define BATCH 2
#define SEQ   2048
#define NEGBIG (-1.0e9f)

// ---------------- WMMA fragment helpers (CDNA5 16x16x32 bf16) ----------------
// A fragment (16 rows x 32 K): lane r=lane&15 is row M, half=lane>>4 selects
// K sub-block {half*8..half*8+7} and {16+half*8..}. Two contiguous 16B loads.
__device__ __forceinline__ v16bf load_a_frag(const bf16_t* base, int lda) {
  const int lane = threadIdx.x & 31;
  const int r    = lane & 15;
  const int half = lane >> 4;
  const bf16_t* p = base + (long)r * lda + half * 8;
  union { v16bf v; v8bf h[2]; } u;
  u.h[0] = *(const v8bf*)p;
  u.h[1] = *(const v8bf*)(p + 16);
  return u.v;
}
// B fragment (32 K x 16 N): lane n=lane&15 is column N, half=lane>>4 selects
// K sub-block {half*8..half*8+7} and {16+half*8..} (strided loads from [K,N]).
__device__ __forceinline__ v16bf load_b_frag(const bf16_t* baseKN, int ldn) {
  const int lane = threadIdx.x & 31;
  const int n    = lane & 15;
  const int half = lane >> 4;
  const bf16_t* p = baseKN + (long)(half * 8) * ldn + n;
  v16bf v;
#pragma unroll
  for (int i = 0; i < 8; ++i) {
    v[i]     = p[(long)i * ldn];
    v[8 + i] = p[(long)(16 + i) * ldn];
  }
  return v;
}
__device__ __forceinline__ v8f wmma_bf16(v16bf a, v16bf b, v8f c) {
  return __builtin_amdgcn_wmma_f32_16x16x32_bf16(false, a, false, b, (short)0, c, false, false);
}

// ---------------- weight convert + transpose: W[N,K] f32 -> Wt[K,N] bf16 ----
__global__ void cvt_transpose_kernel(const float* __restrict__ W, bf16_t* __restrict__ Wt,
                                     int N, int K, long total) {
  long idx = (long)blockIdx.x * blockDim.x + threadIdx.x;
  if (idx >= total) return;
  long nk  = (long)N * K;
  int mat  = (int)(idx / nk);
  long rem = idx - (long)mat * nk;
  int n = (int)(rem / K);
  int k = (int)(rem - (long)n * K);
  Wt[(long)mat * nk + (long)k * N + n] = (bf16_t)W[idx];
}

// ---------------- cumsum of attention mask per batch ----------------
__global__ void cums_kernel(const int* __restrict__ attm, int* __restrict__ cums) {
  int b = blockIdx.x * blockDim.x + threadIdx.x;
  if (b >= BATCH) return;
  int c = 0;
  for (int s = 0; s < SEQ; ++s) { c += attm[b * SEQ + s]; cums[b * SEQ + s] = c; }
}

// ---------------- RMSNorm (f32 in -> bf16 out, global-row indexed out) ------
__global__ void __launch_bounds__(256)
rmsnorm_kernel(const float* __restrict__ x, int globalIn, const float* __restrict__ w,
               bf16_t* __restrict__ outp, int segOff, int Si) {
  __shared__ float red[256];
  const int rid = blockIdx.x;
  const int b = rid / Si;
  const int s = rid - b * Si;
  const int grow = b * SEQ + segOff + s;
  const float* xr = x + (long)(globalIn ? grow : rid) * HID;
  float ss = 0.f;
  for (int i = threadIdx.x; i < HID; i += 256) { float t = xr[i]; ss += t * t; }
  red[threadIdx.x] = ss;
  __syncthreads();
  for (int st = 128; st > 0; st >>= 1) {
    if (threadIdx.x < st) red[threadIdx.x] += red[threadIdx.x + st];
    __syncthreads();
  }
  const float rstd = rsqrtf(red[0] / (float)HID + 1e-6f);
  for (int i = threadIdx.x; i < HID; i += 256)
    outp[(long)grow * HID + i] = (bf16_t)(xr[i] * rstd * w[i]);
}

// ---------------- generic GEMM: C = A(tokens,K) @ Wt(K,N) [+bias|+resid|+=] -
// block = 256 thr (8 waves); wave tile 32x64 (8 wmma / 32-K step); block 128x128
__global__ void __launch_bounds__(256)
gemm_kernel(const bf16_t* __restrict__ A, const bf16_t* __restrict__ Wt,
            const float* __restrict__ bias, const float* __restrict__ resid,
            float* __restrict__ C, int K, int N, int segOff, int Si, int mode) {
  const int tid  = threadIdx.x;
  const int wid  = tid >> 5;
  const int lane = tid & 31;
  const int wrow = wid & 3;
  const int wcol = wid >> 2;
  const int tileM = blockIdx.y * 128 + wrow * 32;
  const int tileN = blockIdx.x * 128 + wcol * 64;
  const int b = tileM / Si;
  const int sBase = tileM - b * Si;
  const long grow0 = (long)b * SEQ + segOff + sBase;
  const bf16_t* Abase = A + grow0 * K;

  v8f acc[2][4];
#pragma unroll
  for (int i = 0; i < 2; ++i)
#pragma unroll
    for (int j = 0; j < 4; ++j)
#pragma unroll
      for (int v = 0; v < 8; ++v) acc[i][j][v] = 0.0f;

  for (int k = 0; k < K; k += 32) {
    v16bf a0 = load_a_frag(Abase + k, K);
    v16bf a1 = load_a_frag(Abase + (long)16 * K + k, K);
#pragma unroll
    for (int j = 0; j < 4; ++j) {
      v16bf bw = load_b_frag(Wt + (long)k * N + tileN + j * 16, N);
      acc[0][j] = wmma_bf16(a0, bw, acc[0][j]);
      acc[1][j] = wmma_bf16(a1, bw, acc[1][j]);
    }
  }

  const int cn = lane & 15;
  const int halfm = (lane >> 4) * 8;
#pragma unroll
  for (int sub = 0; sub < 2; ++sub) {
#pragma unroll
    for (int j = 0; j < 4; ++j) {
      const int n = tileN + j * 16 + cn;
#pragma unroll
      for (int v = 0; v < 8; ++v) {
        const int m = sub * 16 + halfm + v;
        const long row = grow0 + m;
        float val = acc[sub][j][v];
        if (mode == 0)      C[row * N + n] = val + bias[n];
        else if (mode == 1) C[row * N + n] = val + resid[(long)(tileM + m) * N + n];
        else                C[row * N + n] += val;
      }
    }
  }
}

// ---------------- fused gate/up MLP GEMM: silu(A@Gt) * (A@Ut) -> bf16 -------
__global__ void __launch_bounds__(256)
gateup_kernel(const bf16_t* __restrict__ A, const bf16_t* __restrict__ Gt,
              const bf16_t* __restrict__ Ut, bf16_t* __restrict__ OutGU,
              int segOff, int Si) {
  const int K = HID, N = INTER;
  const int tid  = threadIdx.x;
  const int wid  = tid >> 5;
  const int lane = tid & 31;
  const int wrow = wid & 3;
  const int wcol = wid >> 2;
  const int tileM = blockIdx.y * 128 + wrow * 32;
  const int tileN = blockIdx.x * 128 + wcol * 64;
  const int b = tileM / Si;
  const int sBase = tileM - b * Si;
  const long grow0 = (long)b * SEQ + segOff + sBase;
  const bf16_t* Abase = A + grow0 * K;

  v8f accg[2][4], accu[2][4];
#pragma unroll
  for (int i = 0; i < 2; ++i)
#pragma unroll
    for (int j = 0; j < 4; ++j)
#pragma unroll
      for (int v = 0; v < 8; ++v) { accg[i][j][v] = 0.0f; accu[i][j][v] = 0.0f; }

  for (int k = 0; k < K; k += 32) {
    v16bf a0 = load_a_frag(Abase + k, K);
    v16bf a1 = load_a_frag(Abase + (long)16 * K + k, K);
#pragma unroll
    for (int j = 0; j < 4; ++j) {
      v16bf bg = load_b_frag(Gt + (long)k * N + tileN + j * 16, N);
      v16bf bu = load_b_frag(Ut + (long)k * N + tileN + j * 16, N);
      accg[0][j] = wmma_bf16(a0, bg, accg[0][j]);
      accg[1][j] = wmma_bf16(a1, bg, accg[1][j]);
      accu[0][j] = wmma_bf16(a0, bu, accu[0][j]);
      accu[1][j] = wmma_bf16(a1, bu, accu[1][j]);
    }
  }

  const int cn = lane & 15;
  const int halfm = (lane >> 4) * 8;
#pragma unroll
  for (int sub = 0; sub < 2; ++sub) {
#pragma unroll
    for (int j = 0; j < 4; ++j) {
      const int n = tileN + j * 16 + cn;
#pragma unroll
      for (int v = 0; v < 8; ++v) {
        const int m = sub * 16 + halfm + v;
        const long row = grow0 + m;
        float g = accg[sub][j][v];
        float u = accu[sub][j][v];
        float act = g / (1.0f + expf(-g));
        OutGU[row * (long)N + n] = (bf16_t)(act * u);
      }
    }
  }
}

// ---------------- RoPE + pack: q -> [B,NH,S,HD] bf16; k -> [B,NKV,HD,S] bf16
__global__ void rope_qk_kernel(const float* __restrict__ qtmp, const float* __restrict__ ktmp,
                               const int* __restrict__ posids,
                               bf16_t* __restrict__ q_bf, bf16_t* __restrict__ k_t) {
  long idx = (long)blockIdx.x * blockDim.x + threadIdx.x;
  const long total = (long)BATCH * SEQ * (NH + NKV) * 32;
  if (idx >= total) return;
  int j = (int)(idx & 31);
  long t = idx >> 5;
  int head = (int)(t % (NH + NKV));
  long t2 = t / (NH + NKV);
  int s = (int)(t2 % SEQ);
  int b = (int)(t2 / SEQ);
  float pos = (float)posids[b * SEQ + s];
  float invf = powf(1.0e6f, -(float)j * (1.0f / 32.0f));
  float ang = pos * invf;
  float c = cosf(ang), sn = sinf(ang);
  if (head < NH) {
    const float* src = qtmp + ((long)b * SEQ + s) * HID + head * HD;
    float x1 = src[j], x2 = src[j + 32];
    bf16_t* dst = q_bf + (((long)(b * NH + head)) * SEQ + s) * HD;
    dst[j]      = (bf16_t)(x1 * c - x2 * sn);
    dst[j + 32] = (bf16_t)(x2 * c + x1 * sn);
  } else {
    int kv = head - NH;
    const float* src = ktmp + ((long)b * SEQ + s) * (NKV * HD) + kv * HD;
    float x1 = src[j], x2 = src[j + 32];
    bf16_t* dst = k_t + ((long)(b * NKV + kv)) * HD * SEQ;
    dst[(long)j * SEQ + s]        = (bf16_t)(x1 * c - x2 * sn);
    dst[(long)(j + 32) * SEQ + s] = (bf16_t)(x2 * c + x1 * sn);
  }
}

__global__ void packv_kernel(const float* __restrict__ vtmp, bf16_t* __restrict__ v_bf) {
  long idx = (long)blockIdx.x * blockDim.x + threadIdx.x;
  const long total = (long)BATCH * SEQ * NKV * HD;
  if (idx >= total) return;
  int d  = (int)(idx & 63);
  long t = idx >> 6;
  int kv = (int)(t & (NKV - 1));
  long t2 = t >> 1;
  int s = (int)(t2 % SEQ);
  int b = (int)(t2 / SEQ);
  v_bf[(((long)(b * NKV + kv)) * SEQ + s) * HD + d] = (bf16_t)vtmp[idx];
}

// ---------------- flash attention: wave = 16 q rows; 32-key steps -----------
__global__ void __launch_bounds__(128)
flash_kernel(const bf16_t* __restrict__ q_bf, const bf16_t* __restrict__ k_t,
             const bf16_t* __restrict__ v_bf, const int* __restrict__ cums,
             const unsigned char* __restrict__ pad, bf16_t* __restrict__ att) {
  __shared__ int s_cums[SEQ];
  __shared__ unsigned char s_pad[SEQ];
  __shared__ __align__(16) bf16_t s_p[4][16 * 32];

  const int tid  = threadIdx.x;
  const int wid  = tid >> 5;
  const int lane = tid & 31;
  const int b = blockIdx.z;
  const int h = blockIdx.y;
  const int kvh = h / (NH / NKV);

  for (int i = tid; i < SEQ; i += 128) {
    s_cums[i] = cums[b * SEQ + i];
    s_pad[i]  = pad[b * SEQ + i];
  }
  __syncthreads();

  const int qbase = blockIdx.x * 64 + wid * 16;
  const bf16_t* Qbase = q_bf + (((long)(b * NH + h)) * SEQ + qbase) * HD;
  const bf16_t* Kb = k_t + ((long)(b * NKV + kvh)) * HD * SEQ;  // [hd][s]
  const bf16_t* Vb = v_bf + ((long)(b * NKV + kvh)) * SEQ * HD; // [s][hd]

  v16bf qf0 = load_a_frag(Qbase + 0, HD);
  v16bf qf1 = load_a_frag(Qbase + 32, HD);

  const int cn = lane & 15;
  const int halfm = (lane >> 4) * 8;

  float mrow[8], lrow[8];
  int qc[8]; unsigned char qp[8];
  v8f acc[4];
#pragma unroll
  for (int v = 0; v < 8; ++v) {
    mrow[v] = -1.0e30f; lrow[v] = 0.0f;
    int qrow = qbase + halfm + v;
    qc[v] = s_cums[qrow]; qp[v] = s_pad[qrow];
  }
#pragma unroll
  for (int j = 0; j < 4; ++j)
#pragma unroll
    for (int v = 0; v < 8; ++v) acc[j][v] = 0.0f;

  for (int kt = 0; kt < SEQ; kt += 32) {
    v8f sf[2];
#pragma unroll
    for (int nt = 0; nt < 2; ++nt) {
      v8f s = {0.f, 0.f, 0.f, 0.f, 0.f, 0.f, 0.f, 0.f};
      v16bf kf0 = load_b_frag(Kb + (kt + nt * 16), SEQ);
      v16bf kf1 = load_b_frag(Kb + (long)32 * SEQ + (kt + nt * 16), SEQ);
      s = wmma_bf16(qf0, kf0, s);
      s = wmma_bf16(qf1, kf1, s);
      const int key = kt + nt * 16 + cn;
      const int kc = s_cums[key];
      const unsigned char kp = s_pad[key];
#pragma unroll
      for (int v = 0; v < 8; ++v) {
        bool ok = kp && qp[v] && (kc <= qc[v]);
        s[v] = s[v] * 0.125f + (ok ? 0.0f : NEGBIG);
      }
      sf[nt] = s;
    }
#pragma unroll
    for (int v = 0; v < 8; ++v) {
      float rm = fmaxf(sf[0][v], sf[1][v]);
      rm = fmaxf(rm, __shfl_xor(rm, 1, 32));
      rm = fmaxf(rm, __shfl_xor(rm, 2, 32));
      rm = fmaxf(rm, __shfl_xor(rm, 4, 32));
      rm = fmaxf(rm, __shfl_xor(rm, 8, 32));
      float nm = fmaxf(mrow[v], rm);
      float corr = expf(mrow[v] - nm);
      float p0 = expf(sf[0][v] - nm);
      float p1 = expf(sf[1][v] - nm);
      float rs = p0 + p1;
      rs += __shfl_xor(rs, 1, 32);
      rs += __shfl_xor(rs, 2, 32);
      rs += __shfl_xor(rs, 4, 32);
      rs += __shfl_xor(rs, 8, 32);
      lrow[v] = lrow[v] * corr + rs;
      mrow[v] = nm;
#pragma unroll
      for (int j = 0; j < 4; ++j) acc[j][v] = acc[j][v] * corr;
      const int mr = halfm + v;
      s_p[wid][mr * 32 + cn]      = (bf16_t)p0;
      s_p[wid][mr * 32 + 16 + cn] = (bf16_t)p1;
    }
    __syncthreads();
    v16bf pf = load_a_frag(&s_p[wid][0], 32);
#pragma unroll
    for (int j = 0; j < 4; ++j) {
      v16bf vf = load_b_frag(Vb + (long)kt * HD + j * 16, HD);
      acc[j] = wmma_bf16(pf, vf, acc[j]);
    }
    __syncthreads();
  }

#pragma unroll
  for (int j = 0; j < 4; ++j)
#pragma unroll
    for (int v = 0; v < 8; ++v) {
      float o = acc[j][v] / lrow[v];
      long row = (long)b * SEQ + qbase + halfm + v;
      att[row * (NH * HD) + h * HD + j * 16 + cn] = (bf16_t)o;
    }
}

// ---------------- host launch ----------------
extern "C" void kernel_launch(void* const* d_in, const int* in_sizes, int n_in,
                              void* d_out, int out_size, void* d_ws, size_t ws_size,
                              hipStream_t stream) {
  (void)in_sizes; (void)n_in; (void)out_size; (void)ws_size;
  const float* x[3] = {(const float*)d_in[0], (const float*)d_in[1], (const float*)d_in[2]};
  const float* qW  = (const float*)d_in[3];
  const float* qb  = (const float*)d_in[4];
  const float* kW  = (const float*)d_in[5];
  const float* kb  = (const float*)d_in[6];
  const float* vW  = (const float*)d_in[7];
  const float* vb  = (const float*)d_in[8];
  const float* oW  = (const float*)d_in[9];
  const float* ln1 = (const float*)d_in[10];
  const float* ln2 = (const float*)d_in[11];
  const float* gW  = (const float*)d_in[12];
  const float* uW  = (const float*)d_in[13];
  const float* dW  = (const float*)d_in[14];
  const unsigned char* pad = (const unsigned char*)d_in[15];
  const int* attm   = (const int*)d_in[16];
  const int* posids = (const int*)d_in[17];
  float* out = (float*)d_out;

  char* ws = (char*)d_ws;
  size_t off = 0;
  auto take = [&](size_t bytes) -> char* {
    char* p = ws + off;
    off = (off + bytes + 255) & ~(size_t)255;
    return p;
  };
  const long nQW = 3L * HID * HID, nKW = 3L * (NKV * HD) * HID, nGW = 3L * INTER * HID;
  bf16_t* qWt = (bf16_t*)take(nQW * 2);
  bf16_t* kWt = (bf16_t*)take(nKW * 2);
  bf16_t* vWt = (bf16_t*)take(nKW * 2);
  bf16_t* oWt = (bf16_t*)take(nQW * 2);
  bf16_t* gWt = (bf16_t*)take(nGW * 2);
  bf16_t* uWt = (bf16_t*)take(nGW * 2);
  bf16_t* dWt = (bf16_t*)take(nGW * 2);
  const long nTok = (long)BATCH * SEQ;
  bf16_t* h1   = (bf16_t*)take(nTok * HID * 2);
  float*  qtmp = (float*)take(nTok * HID * 4);
  float*  ktmp = (float*)take(nTok * NKV * HD * 4);
  float*  vtmp = (float*)take(nTok * NKV * HD * 4);
  bf16_t* q_bf = (bf16_t*)take(nTok * NH * HD * 2);
  bf16_t* k_t  = (bf16_t*)take(nTok * NKV * HD * 2);
  bf16_t* v_bf = (bf16_t*)take(nTok * NKV * HD * 2);
  bf16_t* attb = (bf16_t*)take(nTok * NH * HD * 2);
  bf16_t* h2   = (bf16_t*)take(nTok * HID * 2);
  bf16_t* gu   = (bf16_t*)take(nTok * INTER * 2);
  int*    cums = (int*)take(nTok * 4);

  auto cdiv = [](long a, long b) -> int { return (int)((a + b - 1) / b); };
  const int T = 256;

  cvt_transpose_kernel<<<cdiv(nQW, T), T, 0, stream>>>(qW, qWt, HID, HID, nQW);
  cvt_transpose_kernel<<<cdiv(nKW, T), T, 0, stream>>>(kW, kWt, NKV * HD, HID, nKW);
  cvt_transpose_kernel<<<cdiv(nKW, T), T, 0, stream>>>(vW, vWt, NKV * HD, HID, nKW);
  cvt_transpose_kernel<<<cdiv(nQW, T), T, 0, stream>>>(oW, oWt, HID, NH * HD, nQW);
  cvt_transpose_kernel<<<cdiv(nGW, T), T, 0, stream>>>(gW, gWt, INTER, HID, nGW);
  cvt_transpose_kernel<<<cdiv(nGW, T), T, 0, stream>>>(uW, uWt, INTER, HID, nGW);
  cvt_transpose_kernel<<<cdiv(nGW, T), T, 0, stream>>>(dW, dWt, HID, INTER, nGW);
  cums_kernel<<<1, 32, 0, stream>>>(attm, cums);

  const int SiA[3]  = {1536, 384, 128};
  const int off3[3] = {0, 1536, 1920};

  // RMSNorm(ln1) + QKV projections
  for (int seg = 0; seg < 3; ++seg) {
    const int Si = SiA[seg], so = off3[seg], M = BATCH * Si;
    rmsnorm_kernel<<<M, 256, 0, stream>>>(x[seg], 0, ln1 + seg * HID, h1, so, Si);
    gemm_kernel<<<dim3(HID / 128, M / 128), 256, 0, stream>>>(
        h1, qWt + (long)seg * HID * HID, qb + seg * HID, nullptr, qtmp, HID, HID, so, Si, 0);
    gemm_kernel<<<dim3(1, M / 128), 256, 0, stream>>>(
        h1, kWt + (long)seg * HID * NKV * HD, kb + seg * NKV * HD, nullptr, ktmp, HID, NKV * HD, so, Si, 0);
    gemm_kernel<<<dim3(1, M / 128), 256, 0, stream>>>(
        h1, vWt + (long)seg * HID * NKV * HD, vb + seg * NKV * HD, nullptr, vtmp, HID, NKV * HD, so, Si, 0);
  }

  const long nrope = (long)BATCH * SEQ * (NH + NKV) * 32;
  rope_qk_kernel<<<cdiv(nrope, T), T, 0, stream>>>(qtmp, ktmp, posids, q_bf, k_t);
  const long nv = (long)BATCH * SEQ * NKV * HD;
  packv_kernel<<<cdiv(nv, T), T, 0, stream>>>(vtmp, v_bf);

  flash_kernel<<<dim3(SEQ / 64, NH, BATCH), 128, 0, stream>>>(q_bf, k_t, v_bf, cums, pad, attb);

  // O-proj + residual, RMSNorm(ln2), gate/up, down-proj accumulate
  for (int seg = 0; seg < 3; ++seg) {
    const int Si = SiA[seg], so = off3[seg], M = BATCH * Si;
    gemm_kernel<<<dim3(HID / 128, M / 128), 256, 0, stream>>>(
        attb, oWt + (long)seg * HID * HID, nullptr, x[seg], out, NH * HD, HID, so, Si, 1);
    rmsnorm_kernel<<<M, 256, 0, stream>>>(out, 1, ln2 + seg * HID, h2, so, Si);
    gateup_kernel<<<dim3(INTER / 128, M / 128), 256, 0, stream>>>(
        h2, gWt + (long)seg * HID * INTER, uWt + (long)seg * HID * INTER, gu, so, Si);
    gemm_kernel<<<dim3(HID / 128, M / 128), 256, 0, stream>>>(
        gu, dWt + (long)seg * INTER * HID, nullptr, nullptr, out, INTER, HID, so, Si, 2);
  }
}
